// Discriminator_8950711845752
// MI455X (gfx1250) — matrix operand order repaired and checked by reading, not compiled, not run on hardware
//
#include <hip/hip_runtime.h>
#include <hip/hip_bf16.h>
#include <math.h>

// ---------------------------------------------------------------------------
// MI455X (gfx1250) implementation. All matrix-heavy work (convs as implicit
// GEMM, GRU input GEMM, GRU recurrent gate GEMMs) goes through
// v_wmma_f32_16x16x32_f16 (f16 operands, f32 accumulate). wave32 everywhere.
// Conv/GEMM use 32x128 block tiles: each wave owns one 16x16 M-tile x two
// N-tiles so the A fragment is reused across two WMMAs. LDS tiles use a
// 40-half row stride so all fragment segments are 16B-aligned (ds_load_b128)
// and bank-conflict-free (20-bank row stride).
// Workspace requirement: ~137M floats (~550 MB).
// ---------------------------------------------------------------------------

typedef __attribute__((ext_vector_type(16))) _Float16 v16h;
typedef __attribute__((ext_vector_type(8)))  _Float16 v8h;
typedef __attribute__((ext_vector_type(8)))  float    v8f;

#define LRELU(x) ((x) > 0.f ? (x) : 0.2f * (x))

static __device__ __forceinline__ float sigm(float x) { return 1.f / (1.f + expf(-x)); }

// WMMA fragment maps used here (wave32; contiguous k is this file's choice, same for A and B):
//  A 16x32 f16 : lane l (row = l&15, lh = l>>4), element e -> K = (lh<<4)+e
//                -> two contiguous 8-half segments at K = lh*16 and K = lh*16+8
//  B 32x16 f16 : lane l (col = l&15), element e -> K = (lh<<4)+e  (16 contiguous halfs)
//  C/D 16x16 f32: element v -> M = (lh<<3)+v, N = l&15

static __device__ __forceinline__ v16h load_a_frag(const _Float16* base /*row ptr*/, int lh) {
  v8h a0 = *(const v8h*)(base + (lh << 4));
  v8h a1 = *(const v8h*)(base + (lh << 4) + 8);
  return __builtin_shufflevector(a0, a1, 0, 1, 2, 3, 4, 5, 6, 7, 8, 9, 10, 11, 12, 13, 14, 15);
}
static __device__ __forceinline__ v16h load_b_frag(const _Float16* base /*col ptr*/, int lh) {
  v8h b0 = *(const v8h*)(base + (lh << 4));
  v8h b1 = *(const v8h*)(base + (lh << 4) + 8);
  return __builtin_shufflevector(b0, b1, 0, 1, 2, 3, 4, 5, 6, 7, 8, 9, 10, 11, 12, 13, 14, 15);
}

// ---------------------------------------------------------------------------
// f32 -> f16 conversion (weight pre-pass)
// ---------------------------------------------------------------------------
__global__ void k_f32_to_f16(const float* __restrict__ s, _Float16* __restrict__ d, int n) {
  int i = blockIdx.x * 256 + threadIdx.x;
  if (i < n) d[i] = (_Float16)s[i];
}

// ---------------------------------------------------------------------------
// Implicit-GEMM conv (NCHW, OIHW weights). Block tile 32(M-pixels) x 128(Cout),
// 8 waves: wave owns 16 M-rows x 32 Cout (two 16x16 WMMA tiles sharing A frag).
// K consumed 32 at a time via LDS im2col staging.
// ---------------------------------------------------------------------------
__global__ __launch_bounds__(256) void k_conv_wmma(
    const float* __restrict__ in, const _Float16* __restrict__ w16,
    const float* __restrict__ bias, float* __restrict__ out,
    int N, int Cin, int H, int W, int Cout, int OH, int OW,
    int KS, int stride, int pad, int K, int fuse_lrelu) {
  __shared__ _Float16 At[32 * 40];
  __shared__ _Float16 Bt[128 * 40];
  const int tid = threadIdx.x;
  const int wave = tid >> 5, lane = tid & 31;
  const int lh = lane >> 4, ll = lane & 15;
  const int m_off = (wave >> 2) * 16;      // 0 or 16
  const int n_off = (wave & 3) * 32;       // 0,32,64,96
  const int mb = blockIdx.x * 32, nb = blockIdx.y * 128;
  const int OHW = OH * OW;
  const int M = N * OHW;
  const int KK = KS * KS;
  const int kk = tid & 31;

  int nimg[4], ihb[4], iwb[4], valid[4];
#pragma unroll
  for (int i = 0; i < 4; ++i) {
    int m = mb + wave + 8 * i;
    valid[i] = (m < M);
    int mm = valid[i] ? m : 0;
    int ni = mm / OHW;
    int rem = mm - ni * OHW;
    int oh = rem / OW, ow = rem - oh * OW;
    nimg[i] = ni;
    ihb[i] = oh * stride - pad;
    iwb[i] = ow * stride - pad;
  }

  v8f acc0 = {}, acc1 = {};
  for (int k0 = 0; k0 < K; k0 += 32) {
    int k = k0 + kk;
    int ci = k / KK;
    int r2 = k - ci * KK;
    int kh = r2 / KS, kw = r2 - kh * KS;
#pragma unroll
    for (int i = 0; i < 4; ++i) {   // A: 32 rows x 32 k
      float v = 0.f;
      if (k < K && valid[i]) {
        int ih = ihb[i] + kh, iw = iwb[i] + kw;
        if (ih >= 0 && ih < H && iw >= 0 && iw < W)
          v = in[((size_t)(nimg[i] * Cin + ci) * H + ih) * W + iw];
      }
      At[(wave + 8 * i) * 40 + kk] = (_Float16)v;
    }
#pragma unroll
    for (int i = 0; i < 16; ++i) {  // B: 128 rows x 32 k
      int co = nb + wave + 8 * i;
      _Float16 v = (_Float16)0.f;
      if (co < Cout && k < K) v = w16[(size_t)co * K + k];
      Bt[(wave + 8 * i) * 40 + kk] = v;
    }
    if (k0 + 32 < K)  // hint: pull next weight chunk toward cache (global_prefetch_b8)
      __builtin_prefetch(&w16[(size_t)(nb + wave * 16) * K + k0 + 32], 0, 1);
    __syncthreads();
    v16h a = load_a_frag(At + (m_off + ll) * 40, lh);
    v16h b0 = load_b_frag(Bt + (n_off + ll) * 40, lh);
    v16h b1 = load_b_frag(Bt + (n_off + 16 + ll) * 40, lh);
    acc0 = __builtin_amdgcn_wmma_f32_16x16x32_f16(false, a, false, b0, (short)0, acc0, false, false);
    acc1 = __builtin_amdgcn_wmma_f32_16x16x32_f16(false, a, false, b1, (short)0, acc1, false, false);
    __syncthreads();
  }
#pragma unroll
  for (int t = 0; t < 2; ++t) {
    v8f acc = t ? acc1 : acc0;
#pragma unroll
    for (int v = 0; v < 8; ++v) {
      int m = mb + m_off + (lh << 3) + v;
      int co = nb + n_off + t * 16 + ll;
      if (m < M && co < Cout) {
        int ni = m / OHW, rem = m - ni * OHW;
        float val = acc[v];
        if (bias) val += bias[co];
        if (fuse_lrelu) val = LRELU(val);
        out[(size_t)(ni * Cout + co) * OHW + rem] = val;
      }
    }
  }
}

// ---------------------------------------------------------------------------
// Dense GEMM: C[M x Nc] = A[M x K] * B[Nc x K]^T (+bias). f16 in, f32 out.
// Same 32x128 tiling as the conv kernel.
// ---------------------------------------------------------------------------
__global__ __launch_bounds__(256) void k_gemm_wmma(
    const _Float16* __restrict__ A16, const _Float16* __restrict__ B16,
    const float* __restrict__ bias, float* __restrict__ C, int M, int Nc, int K) {
  __shared__ _Float16 At[32 * 40];
  __shared__ _Float16 Bt[128 * 40];
  const int tid = threadIdx.x;
  const int wave = tid >> 5, lane = tid & 31;
  const int lh = lane >> 4, ll = lane & 15;
  const int m_off = (wave >> 2) * 16;
  const int n_off = (wave & 3) * 32;
  const int mb = blockIdx.x * 32, nb = blockIdx.y * 128;
  const int kk = tid & 31;

  v8f acc0 = {}, acc1 = {};
  for (int k0 = 0; k0 < K; k0 += 32) {
    int k = k0 + kk;
#pragma unroll
    for (int i = 0; i < 4; ++i) {
      int mr = mb + wave + 8 * i;
      _Float16 v = (_Float16)0.f;
      if (mr < M && k < K) v = A16[(size_t)mr * K + k];
      At[(wave + 8 * i) * 40 + kk] = v;
    }
#pragma unroll
    for (int i = 0; i < 16; ++i) {
      int nr = nb + wave + 8 * i;
      _Float16 v = (_Float16)0.f;
      if (nr < Nc && k < K) v = B16[(size_t)nr * K + k];
      Bt[(wave + 8 * i) * 40 + kk] = v;
    }
    __syncthreads();
    v16h a = load_a_frag(At + (m_off + ll) * 40, lh);
    v16h b0 = load_b_frag(Bt + (n_off + ll) * 40, lh);
    v16h b1 = load_b_frag(Bt + (n_off + 16 + ll) * 40, lh);
    acc0 = __builtin_amdgcn_wmma_f32_16x16x32_f16(false, a, false, b0, (short)0, acc0, false, false);
    acc1 = __builtin_amdgcn_wmma_f32_16x16x32_f16(false, a, false, b1, (short)0, acc1, false, false);
    __syncthreads();
  }
#pragma unroll
  for (int t = 0; t < 2; ++t) {
    v8f acc = t ? acc1 : acc0;
#pragma unroll
    for (int v = 0; v < 8; ++v) {
      int m = mb + m_off + (lh << 3) + v, n = nb + n_off + t * 16 + ll;
      if (m < M && n < Nc) C[(size_t)m * Nc + n] = acc[v] + (bias ? bias[n] : 0.f);
    }
  }
}

// ---------------------------------------------------------------------------
// One GRU step (B=32, H=256). Single workgroup, 8 waves. Each wave owns four
// (m-tile, 16-channel strip) pairs and computes r/z/n gate tiles with 3 WMMA
// accumulators over K=256, then fuses the gate nonlinearity + state update.
// gh = h @ w_hh^T (+ b_hh in epilogue); gi precomputed. B fragments are
// 32B-aligned v16h loads straight from the f16 weight matrix.
// ---------------------------------------------------------------------------
__global__ __launch_bounds__(256) void k_gru_step(
    const float* __restrict__ h_in,        // 32x256 or nullptr (h0 = 0)
    const float* __restrict__ gi,          // 32x768 for this step
    const _Float16* __restrict__ whh,      // 768x256 f16
    const float* __restrict__ bhh,         // 768
    const int* __restrict__ txt_len, int s, int use_mask,
    float* __restrict__ h_store) {         // 32x256 (= hseq[s], chained)
  __shared__ _Float16 hA[32 * 264];
  const int tid = threadIdx.x;
  for (int e = tid; e < 32 * 256; e += 256) {
    int r = e >> 8, c = e & 255;
    float v = h_in ? h_in[e] : 0.f;
    hA[r * 264 + c] = (_Float16)v;
  }
  __syncthreads();
  const int wave = tid >> 5, lane = tid & 31;
  const int lh = lane >> 4, ll = lane & 15;
  for (int q = 0; q < 4; ++q) {
    int p = wave * 4 + q;           // 0..31
    int strip = p >> 1;             // 0..15 -> 16-channel strip
    int m_off = (p & 1) * 16;       // batch tile
    int c0 = strip * 16;
    v8f ar = {}, az = {}, an = {};
    for (int kb = 0; kb < 8; ++kb) {
      v16h a = load_a_frag(hA + (m_off + ll) * 264 + kb * 32, lh);
      int j = c0 + ll, kbb = kb * 32 + (lh << 4);
      v16h br  = *(const v16h*)(whh + (size_t)(j) * 256 + kbb);
      v16h bz  = *(const v16h*)(whh + (size_t)(j + 256) * 256 + kbb);
      v16h bn_ = *(const v16h*)(whh + (size_t)(j + 512) * 256 + kbb);
      ar = __builtin_amdgcn_wmma_f32_16x16x32_f16(false, a, false, br,  (short)0, ar, false, false);
      az = __builtin_amdgcn_wmma_f32_16x16x32_f16(false, a, false, bz,  (short)0, az, false, false);
      an = __builtin_amdgcn_wmma_f32_16x16x32_f16(false, a, false, bn_, (short)0, an, false, false);
    }
#pragma unroll
    for (int v = 0; v < 8; ++v) {
      int m = m_off + (lh << 3) + v;  // batch index
      int c = c0 + ll;                // channel
      float hr = ar[v] + bhh[c];
      float hz = az[v] + bhh[256 + c];
      float hn = an[v] + bhh[512 + c];
      const float* gim = gi + (size_t)m * 768;
      float r = sigm(gim[c] + hr);
      float z = sigm(gim[256 + c] + hz);
      float n = tanhf(gim[512 + c] + r * hn);
      float hold = h_in ? h_in[m * 256 + c] : 0.f;
      float hnew = (1.f - z) * n + z * hold;
      if (use_mask) {
        float mk = (s < txt_len[m]) ? 1.f : 0.f;
        hnew = mk * hnew + (1.f - mk) * hold;
      }
      h_store[m * 256 + c] = hnew;
    }
  }
}

// ---------------------------------------------------------------------------
// BatchNorm stats: per-channel mean + rsqrt(var+1e-5) over (N=32, HW).
// ---------------------------------------------------------------------------
__global__ void k_bn_stats(const float* __restrict__ x, float* __restrict__ mean,
                           float* __restrict__ rstd, int C, int HW) {
  int c = blockIdx.x;
  __shared__ float s1[256];
  __shared__ float s2[256];
  float a1 = 0.f, a2 = 0.f;
  int total = 32 * HW;
  for (int t = threadIdx.x; t < total; t += 256) {
    int n = t / HW, i = t - n * HW;
    float v = x[(size_t)(n * C + c) * HW + i];
    a1 += v;
    a2 += v * v;
  }
  s1[threadIdx.x] = a1;
  s2[threadIdx.x] = a2;
  __syncthreads();
  for (int st = 128; st > 0; st >>= 1) {
    if (threadIdx.x < st) {
      s1[threadIdx.x] += s1[threadIdx.x + st];
      s2[threadIdx.x] += s2[threadIdx.x + st];
    }
    __syncthreads();
  }
  if (threadIdx.x == 0) {
    float mu = s1[0] / total;
    float var = s2[0] / total - mu * mu;
    mean[c] = mu;
    rstd[c] = rsqrtf(var + 1e-5f);
  }
}

__global__ void k_bn_apply_lrelu(float* __restrict__ x, const float* __restrict__ mean,
                                 const float* __restrict__ rstd, const float* __restrict__ g,
                                 const float* __restrict__ be, int C, int HW, int total) {
  int i = blockIdx.x * 256 + threadIdx.x;
  if (i >= total) return;
  int c = (i / HW) % C;
  float v = (x[i] - mean[c]) * rstd[c] * g[c] + be[c];
  x[i] = LRELU(v);
}

// BN + lrelu + spatial mean fused (g-branch pooled features)
__global__ void k_bn_lrelu_mean(const float* __restrict__ x, const float* __restrict__ mean,
                                const float* __restrict__ rstd, const float* __restrict__ g,
                                const float* __restrict__ be, float* __restrict__ pooled,
                                int C, int HW) {
  int c = blockIdx.x, n = blockIdx.y;
  const float* xp = x + (size_t)(n * C + c) * HW;
  __shared__ float s[256];
  float a = 0.f;
  for (int i = threadIdx.x; i < HW; i += 256) {
    float v = (xp[i] - mean[c]) * rstd[c] * g[c] + be[c];
    a += LRELU(v);
  }
  s[threadIdx.x] = a;
  __syncthreads();
  for (int st = 128; st > 0; st >>= 1) {
    if (threadIdx.x < st) s[threadIdx.x] += s[threadIdx.x + st];
    __syncthreads();
  }
  if (threadIdx.x == 0) pooled[n * C + c] = s[0] / (float)HW;
}

__global__ void k_maxpool2(const float* __restrict__ in, float* __restrict__ out,
                           int NC, int H, int W) {
  int OH = H >> 1, OW = W >> 1;
  int i = blockIdx.x * 256 + threadIdx.x;
  int total = NC * OH * OW;
  if (i >= total) return;
  int nc = i / (OH * OW), rem = i - nc * (OH * OW);
  int oh = rem / OW, ow = rem - oh * OW;
  const float* p = in + ((size_t)nc * H + 2 * oh) * W + 2 * ow;
  out[i] = fmaxf(fmaxf(p[0], p[1]), fmaxf(p[W], p[W + 1]));
}

// cls head: D[b] = <f3[b], w> + b0 (512*4*4 = 8192 dot)
__global__ void k_cls(const float* __restrict__ f3, const float* __restrict__ w,
                      const float* __restrict__ b, float* __restrict__ D) {
  int bb = blockIdx.x;
  __shared__ float s[256];
  float a = 0.f;
  const float* p = f3 + (size_t)bb * 8192;
  for (int i = threadIdx.x; i < 8192; i += 256) a += p[i] * w[i];
  s[threadIdx.x] = a;
  __syncthreads();
  for (int st = 128; st > 0; st >>= 1) {
    if (threadIdx.x < st) s[threadIdx.x] += s[threadIdx.x + st];
    __syncthreads();
  }
  if (threadIdx.x == 0) D[bb] = s[0] + b[0];
}

// tokens[s][b][c] = emb[tok] + pos_emb[s]; also f16 copy for the WMMA GEMM
__global__ void k_embed(const int* __restrict__ text, const float* __restrict__ emb,
                        const float* __restrict__ pos, float* __restrict__ tokf,
                        _Float16* __restrict__ tok16) {
  int i = blockIdx.x * 256 + threadIdx.x;
  if (i >= 80 * 32 * 256) return;
  int c = i & 255;
  int sb = i >> 8;
  int b = sb & 31, s = sb >> 5;
  int t = text[b * 80 + s];
  if (t == 0) t = 9920 + s;        // NUM_TOK - SEQ + s
  if (t > 8055) t = 8055;          // JAX gather clips OOB
  if (t < 0) t = 0;
  float v = emb[(size_t)t * 256 + c] + pos[s * 256 + c];
  tokf[i] = v;
  tok16[i] = (_Float16)v;
}

// u = (hf*mask + hb)/2 ; m[b][c] = sum_s u / txt_len[b]
__global__ void k_u_m(const float* __restrict__ hf, const float* __restrict__ hb,
                      const int* __restrict__ txt_len, float* __restrict__ u,
                      float* __restrict__ mvec) {
  int i = blockIdx.x * 256 + threadIdx.x;
  if (i >= 32 * 256) return;
  int b = i >> 8, c = i & 255;
  int len = txt_len[b];
  if (len > 80) len = 80;
  float sum = 0.f;
  for (int s = 0; s < 80; ++s) {
    float mk = (s < len) ? 1.f : 0.f;
    int idx = (s * 32 + b) * 256 + c;
    float uu = (hf[idx] * mk + hb[idx]) * 0.5f;
    u[idx] = uu;
    sum += uu;
  }
  mvec[i] = sum / (float)len;
}

__global__ void k_att(const float* __restrict__ u, const float* __restrict__ mvec,
                      float* __restrict__ att) {
  int i = blockIdx.x * 256 + threadIdx.x;
  if (i >= 2560) return;
  int s = i >> 5, b = i & 31;
  const float* up = u + (size_t)(s * 32 + b) * 256;
  const float* mp = mvec + b * 256;
  float a = 0.f;
  for (int c = 0; c < 256; ++c) a += up[c] * mp[c];
  att[s * 32 + b] = a;
}

__global__ void k_att_softmax(float* __restrict__ att, const int* __restrict__ txt_len) {
  int b = threadIdx.x;
  if (b >= 32) return;
  int len = txt_len[b];
  if (len > 80) len = 80;
  float sum = 0.f;
  for (int s = 0; s < 80; ++s) {
    float e = (s < len) ? expf(att[s * 32 + b]) : 0.f;
    att[s * 32 + b] = e;
    sum += e;
  }
  float inv = 1.f / sum;
  for (int s = 0; s < 80; ++s) att[s * 32 + b] *= inv;
}

// weight[k][s][b] = softmax_k(u[s][b] . gw_w[k] + gw_b[k])
__global__ void k_weight(const float* __restrict__ u, const float* __restrict__ gww,
                         const float* __restrict__ gwb, float* __restrict__ w3) {
  int i = blockIdx.x * 256 + threadIdx.x;
  if (i >= 2560) return;
  int s = i >> 5, b = i & 31;
  const float* up = u + (size_t)(s * 32 + b) * 256;
  float l0 = gwb[0], l1 = gwb[1], l2 = gwb[2];
  for (int c = 0; c < 256; ++c) {
    float v = up[c];
    l0 += v * gww[c];
    l1 += v * gww[256 + c];
    l2 += v * gww[512 + c];
  }
  float mx = fmaxf(l0, fmaxf(l1, l2));
  float e0 = expf(l0 - mx), e1 = expf(l1 - mx), e2 = expf(l2 - mx);
  float inv = 1.f / (e0 + e1 + e2);
  w3[0 * 2560 + s * 32 + b] = e0 * inv;
  w3[1 * 2560 + s * 32 + b] = e1 * inv;
  w3[2 * 2560 + s * 32 + b] = e2 * inv;
}

// v[b][c] = gfw[dim][c] + sum_j pooled[b][j]*gfw[j][c];  s[b] = gfb[dim] + gfb.pooled
__global__ void k_vi_si(const float* __restrict__ gfw, const float* __restrict__ gfb,
                        const float* __restrict__ pooled, int dim,
                        float* __restrict__ vtmp, float* __restrict__ stmp) {
  int i = blockIdx.x * 256 + threadIdx.x;
  if (i >= 32 * 256) return;
  int b = i >> 8, c = i & 255;
  const float* pp = pooled + (size_t)b * dim;
  float a = gfw[(size_t)dim * 256 + c];
  for (int j = 0; j < dim; ++j) a += pp[j] * gfw[(size_t)j * 256 + c];
  vtmp[i] = a;
  if (c == 0) {
    float sb = gfb[dim];
    for (int j = 0; j < dim; ++j) sb += gfb[j] * pp[j];
    stmp[b] = sb;
  }
}

// simacc[s][b] (+)= sigmoid(u[s][b].v[b] + s[b]) * w3[k][s][b]
__global__ void k_sim_accum(const float* __restrict__ u, const float* __restrict__ vtmp,
                            const float* __restrict__ stmp, const float* __restrict__ w3,
                            float* __restrict__ simacc, int k, int first) {
  int i = blockIdx.x * 256 + threadIdx.x;
  if (i >= 2560) return;
  int s = i >> 5, b = i & 31;
  const float* up = u + (size_t)(s * 32 + b) * 256;
  const float* vp = vtmp + b * 256;
  float x = stmp[b];
  for (int c = 0; c < 256; ++c) x += up[c] * vp[c];
  float contrib = sigm(x) * w3[k * 2560 + s * 32 + b];
  simacc[s * 32 + b] = first ? contrib : simacc[s * 32 + b] + contrib;
}

// sim[b] = prod_s min(simacc+eps,1)^att = exp(sum_s att*log(min(...)))
__global__ void k_sim_final(const float* __restrict__ simacc, const float* __restrict__ att,
                            float* __restrict__ out_sim) {
  int b = threadIdx.x;
  if (b >= 32) return;
  float acc = 0.f;
  for (int s = 0; s < 80; ++s) {
    float v = fminf(simacc[s * 32 + b] + 1e-7f, 1.f);
    acc += att[s * 32 + b] * logf(v);
  }
  out_sim[b] = expf(acc);
}

// ---------------------------------------------------------------------------
// Host orchestration
// ---------------------------------------------------------------------------
extern "C" void kernel_launch(void* const* d_in, const int* in_sizes, int n_in,
                              void* d_out, int out_size, void* d_ws, size_t ws_size,
                              hipStream_t stream) {
  (void)in_sizes; (void)n_in; (void)out_size; (void)ws_size;

  const float* img    = (const float*)d_in[0];
  const int*   text   = (const int*)d_in[1];
  const int*   tlen   = (const int*)d_in[2];
  const float* e1_w1  = (const float*)d_in[3];
  const float* e1_b1  = (const float*)d_in[4];
  const float* e1_w2  = (const float*)d_in[5];
  const float* e1_g   = (const float*)d_in[6];
  const float* e1_be  = (const float*)d_in[7];
  const float* e2_w   = (const float*)d_in[8];
  const float* e2_g   = (const float*)d_in[9];
  const float* e2_be  = (const float*)d_in[10];
  const float* e3_w   = (const float*)d_in[11];
  const float* e3_g   = (const float*)d_in[12];
  const float* e3_be  = (const float*)d_in[13];
  const float* g1_w   = (const float*)d_in[14];
  const float* g1_g   = (const float*)d_in[15];
  const float* g1_be  = (const float*)d_in[16];
  const float* g2_w   = (const float*)d_in[17];
  const float* g2_g   = (const float*)d_in[18];
  const float* g2_be  = (const float*)d_in[19];
  const float* g3_w   = (const float*)d_in[20];
  const float* g3_g   = (const float*)d_in[21];
  const float* g3_be  = (const float*)d_in[22];
  const float* cls_w  = (const float*)d_in[23];
  const float* cls_b  = (const float*)d_in[24];
  const float* emb    = (const float*)d_in[25];
  const float* pos    = (const float*)d_in[26];
  const float* wf_ih  = (const float*)d_in[27];
  const float* wf_hh  = (const float*)d_in[28];
  const float* bf_ih  = (const float*)d_in[29];
  const float* bf_hh  = (const float*)d_in[30];
  const float* wb_ih  = (const float*)d_in[31];
  const float* wb_hh  = (const float*)d_in[32];
  const float* bb_ih  = (const float*)d_in[33];
  const float* bb_hh  = (const float*)d_in[34];
  const float* gf0_w  = (const float*)d_in[35];
  const float* gf0_b  = (const float*)d_in[36];
  const float* gf1_w  = (const float*)d_in[37];
  const float* gf1_b  = (const float*)d_in[38];
  const float* gf2_w  = (const float*)d_in[39];
  const float* gf2_b  = (const float*)d_in[40];
  const float* gw_w   = (const float*)d_in[41];
  const float* gw_b   = (const float*)d_in[42];
  float* out = (float*)d_out;

  // -------- workspace layout (floats); total ~136.6M floats (~547 MB) -------
  float* wsf = (float*)d_ws;
  size_t off = 0;
  auto take = [&](size_t n) { float* p = wsf + off; off += n; return p; };
  float* bufA  = take(67108864);  // conv1 out (32,128,128,128); reused for g1r/g2r
  float* f1    = take(33554432);  // (32,256,64,64)
  float* f2    = take(16777216);  // (32,512,32,32)
  float* f2p   = take(4194304);   // (32,512,16,16)
  float* e3r   = take(1048576);   // (32,512,8,8)
  float* f3    = take(262144);    // (32,512,4,4)
  float* g3r   = take(262144);
  float* bnm   = take(512);
  float* bnr   = take(512);
  float* pool0 = take(8192);      // (32,256)
  float* pool1 = take(16384);     // (32,512)
  float* pool2 = take(16384);
  float* tokf  = take(655360);    // (80,32,256)
  _Float16* tok16 = (_Float16*)take(327680);
  float* gi_f  = take(1966080);   // (2560,768)
  float* gi_b  = take(1966080);
  float* hf    = take(655360);    // (80,32,256)
  float* hb    = take(655360);
  float* uu    = take(655360);
  float* mvec  = take(8192);
  float* att   = take(2560);
  float* w3    = take(7680);
  float* vtmp  = take(8192);
  float* stmp  = take(32);
  float* simacc = take(2560);
  _Float16* w16_e1w1 = (_Float16*)take(3072);     // 6144 halfs
  _Float16* w16_e1w2 = (_Float16*)take(262144);   // 524288
  _Float16* w16_e2   = (_Float16*)take(1048576);  // 2097152
  _Float16* w16_e3   = (_Float16*)take(2097152);  // 4194304
  _Float16* w16_g1   = (_Float16*)take(294912);   // 589824
  _Float16* w16_g2   = (_Float16*)take(1179648);  // 2359296
  _Float16* w16_g3   = (_Float16*)take(1179648);
  _Float16* w16_wfih = (_Float16*)take(98304);    // 196608
  _Float16* w16_wfhh = (_Float16*)take(98304);
  _Float16* w16_wbih = (_Float16*)take(98304);
  _Float16* w16_wbhh = (_Float16*)take(98304);
  float* g1r = bufA;                 // reuse after e1_2 consumes conv1 output
  float* g2r = bufA + 33554432;

  auto cvt = [&](const float* s, _Float16* d, int n) {
    k_f32_to_f16<<<(n + 255) / 256, 256, 0, stream>>>(s, d, n);
  };
  cvt(e1_w1, w16_e1w1, 6144);
  cvt(e1_w2, w16_e1w2, 524288);
  cvt(e2_w, w16_e2, 2097152);
  cvt(e3_w, w16_e3, 4194304);
  cvt(g1_w, w16_g1, 589824);
  cvt(g2_w, w16_g2, 2359296);
  cvt(g3_w, w16_g3, 2359296);
  cvt(wf_ih, w16_wfih, 196608);
  cvt(wf_hh, w16_wfhh, 196608);
  cvt(wb_ih, w16_wbih, 196608);
  cvt(wb_hh, w16_wbhh, 196608);

  // ------------------------------ CNN branch -------------------------------
  // conv1: 3->128 k4 s2 p1 + bias + lrelu  (Cout=128 -> one 128-tile)
  k_conv_wmma<<<dim3(524288 / 32, 1), 256, 0, stream>>>(
      img, w16_e1w1, e1_b1, bufA, 32, 3, 256, 256, 128, 128, 128, 4, 2, 1, 48, 1);
  // e1_2: 128->256 k4 s2 p1
  k_conv_wmma<<<dim3(131072 / 32, 2), 256, 0, stream>>>(
      bufA, w16_e1w2, nullptr, f1, 32, 128, 128, 128, 256, 64, 64, 4, 2, 1, 2048, 0);
  k_bn_stats<<<256, 256, 0, stream>>>(f1, bnm, bnr, 256, 4096);
  k_bn_apply_lrelu<<<(33554432 + 255) / 256, 256, 0, stream>>>(f1, bnm, bnr, e1_g, e1_be, 256, 4096, 33554432);
  // e2: 256->512 k4 s2 p1
  k_conv_wmma<<<dim3(32768 / 32, 4), 256, 0, stream>>>(
      f1, w16_e2, nullptr, f2, 32, 256, 64, 64, 512, 32, 32, 4, 2, 1, 4096, 0);
  k_bn_stats<<<512, 256, 0, stream>>>(f2, bnm, bnr, 512, 1024);
  k_bn_apply_lrelu<<<(16777216 + 255) / 256, 256, 0, stream>>>(f2, bnm, bnr, e2_g, e2_be, 512, 1024, 16777216);
  // maxpool f2 -> f2p; e3 conv; maxpool -> f3; bn+lrelu
  k_maxpool2<<<(4194304 + 255) / 256, 256, 0, stream>>>(f2, f2p, 32 * 512, 32, 32);
  k_conv_wmma<<<dim3(2048 / 32, 4), 256, 0, stream>>>(
      f2p, w16_e3, nullptr, e3r, 32, 512, 16, 16, 512, 8, 8, 4, 2, 1, 8192, 0);
  k_maxpool2<<<(262144 + 255) / 256, 256, 0, stream>>>(e3r, f3, 32 * 512, 8, 8);
  k_bn_stats<<<512, 256, 0, stream>>>(f3, bnm, bnr, 512, 16);
  k_bn_apply_lrelu<<<(262144 + 255) / 256, 256, 0, stream>>>(f3, bnm, bnr, e3_g, e3_be, 512, 16, 262144);
  // g1 branch on f1
  k_conv_wmma<<<dim3(131072 / 32, 2), 256, 0, stream>>>(
      f1, w16_g1, nullptr, g1r, 32, 256, 64, 64, 256, 64, 64, 3, 1, 1, 2304, 0);
  k_bn_stats<<<256, 256, 0, stream>>>(g1r, bnm, bnr, 256, 4096);
  k_bn_lrelu_mean<<<dim3(256, 32), 256, 0, stream>>>(g1r, bnm, bnr, g1_g, g1_be, pool0, 256, 4096);
  // g2 branch on f2
  k_conv_wmma<<<dim3(32768 / 32, 4), 256, 0, stream>>>(
      f2, w16_g2, nullptr, g2r, 32, 512, 32, 32, 512, 32, 32, 3, 1, 1, 4608, 0);
  k_bn_stats<<<512, 256, 0, stream>>>(g2r, bnm, bnr, 512, 1024);
  k_bn_lrelu_mean<<<dim3(512, 32), 256, 0, stream>>>(g2r, bnm, bnr, g2_g, g2_be, pool1, 512, 1024);
  // g3 branch on f3
  k_conv_wmma<<<dim3(512 / 32, 4), 256, 0, stream>>>(
      f3, w16_g3, nullptr, g3r, 32, 512, 4, 4, 512, 4, 4, 3, 1, 1, 4608, 0);
  k_bn_stats<<<512, 256, 0, stream>>>(g3r, bnm, bnr, 512, 16);
  k_bn_lrelu_mean<<<dim3(512, 32), 256, 0, stream>>>(g3r, bnm, bnr, g3_g, g3_be, pool2, 512, 16);
  // cls head -> D
  k_cls<<<32, 256, 0, stream>>>(f3, cls_w, cls_b, out);

  // ------------------------------ text branch ------------------------------
  k_embed<<<(655360 + 255) / 256, 256, 0, stream>>>(text, emb, pos, tokf, tok16);
  // gi = tokens @ w_ih^T + b_ih for both directions (hoisted out of the scan)
  k_gemm_wmma<<<dim3(80, 6), 256, 0, stream>>>(tok16, w16_wfih, bf_ih, gi_f, 2560, 768, 256);
  k_gemm_wmma<<<dim3(80, 6), 256, 0, stream>>>(tok16, w16_wbih, bb_ih, gi_b, 2560, 768, 256);
  // forward scan (no masking inside step)
  for (int s = 0; s < 80; ++s) {
    const float* hin = (s == 0) ? nullptr : (hf + (size_t)(s - 1) * 8192);
    k_gru_step<<<1, 256, 0, stream>>>(hin, gi_f + (size_t)s * 32 * 768, w16_wfhh, bf_hh,
                                      tlen, s, 0, hf + (size_t)s * 8192);
  }
  // backward scan with masking; store at original index s
  for (int t = 0; t < 80; ++t) {
    int s = 79 - t;
    const float* hin = (t == 0) ? nullptr : (hb + (size_t)(s + 1) * 8192);
    k_gru_step<<<1, 256, 0, stream>>>(hin, gi_b + (size_t)s * 32 * 768, w16_wbhh, bb_hh,
                                      tlen, s, 1, hb + (size_t)s * 8192);
  }
  // u, m, attention, gate weights
  k_u_m<<<32, 256, 0, stream>>>(hf, hb, tlen, uu, mvec);
  k_att<<<10, 256, 0, stream>>>(uu, mvec, att);
  k_att_softmax<<<1, 32, 0, stream>>>(att, tlen);
  k_weight<<<10, 256, 0, stream>>>(uu, gw_w, gw_b, w3);
  // similarity accumulation over the 3 granularities
  const float* gfw[3] = {gf0_w, gf1_w, gf2_w};
  const float* gfb[3] = {gf0_b, gf1_b, gf2_b};
  const float* pooled[3] = {pool0, pool1, pool2};
  const int dims[3] = {256, 512, 512};
  for (int k = 0; k < 3; ++k) {
    k_vi_si<<<32, 256, 0, stream>>>(gfw[k], gfb[k], pooled[k], dims[k], vtmp, stmp);
    k_sim_accum<<<10, 256, 0, stream>>>(uu, vtmp, stmp, w3, simacc, k, k == 0);
  }
  k_sim_final<<<1, 32, 0, stream>>>(simacc, att, out + 32);
}
